// unit_ctrgcn_16922171146681
// MI455X (gfx1250) — compile-verified, matrix operand order checked by reading, not run on hardware
//
#include <hip/hip_runtime.h>

// CTR-GCN unit for MI455X (gfx1250): bf16 WMMA for the two GEMMs,
// f32 accumulate, LDS-resident attention map, fp32 global atomics for
// the subset reduction. Operand gathers use 32-bit LDS loads (each WMMA
// operand VGPR holds two consecutive K values per ISA 7.12.2).

typedef __attribute__((ext_vector_type(16))) __bf16 v16bf;
typedef __attribute__((ext_vector_type(8)))  float  v8f;

#define NB 256   // batch
#define CC 64    // in channels
#define TT 64    // frames
#define VV 25    // joints
#define SS 3     // subsets
#define RR 8     // rel channels
#define OO 64    // out channels
#define OT 16    // o-tile per block
#define HH 16    // SE hidden
#define BN_EPS 1e-5f

__device__ __forceinline__ unsigned short f2bf(float f) {
  unsigned int u = __float_as_uint(f);
  unsigned int r = u + 0x7FFFu + ((u >> 16) & 1u);   // round-to-nearest-even
  return (unsigned short)(r >> 16);
}

// K base for operand register j (0..7): register j holds K = kpair, kpair+1.
// Per ISA 7.12.2 16-bit A-matrix 16x32 layout (B uses the mirrored mapping).
__device__ __forceinline__ int kpair(int j, int lane) {
  return ((j >= 4) ? 16 : 0) + ((j & 3) << 1) + ((lane >= 16) ? 8 : 0);
}

union BF16x16 {
  unsigned int   w[8];   // each uint = two consecutive-K bf16 values
  v16bf          v;
};

// ---------------------------------------------------------------- prep ----
// xh = bf16(x); xm[n,c,v] = mean_t x[n,c,t,v]
__global__ __launch_bounds__(256) void prep_kernel(
    const float* __restrict__ x, unsigned short* __restrict__ xh,
    float* __restrict__ xm) {
  __shared__ float sumv[VV];
  int tid = threadIdx.x;
  int nc = blockIdx.x;                       // n*CC + c
  if (tid < VV) sumv[tid] = 0.f;
  __syncthreads();
  size_t base = (size_t)nc * (TT * VV);
  for (int i = tid; i < TT * VV; i += 256) {
    float f = x[base + i];
    xh[base + i] = f2bf(f);
    atomicAdd(&sumv[i % VV], f);
  }
  __syncthreads();
  if (tid < VV) xm[(size_t)nc * VV + tid] = sumv[tid] * (1.f / TT);
}

// ---------------------------------------------------------------- main ----
struct MainLds {
  float xm[CC * VV];            // [c][v]
  float x1[RR * VV];
  float x2[RR * VV];
  float pasym[VV * VV];         // PA + PA^T
  float w4l[OT * RR];
  float b4l[OT];
  float b3l[OT];
  float att[RR * VV * VV];      // tanh(x1[r,u]-x2[r,v])
  float msum[OT * VV];          // sum_v m[o,u,v]
  alignas(16) unsigned short w3l[OT * CC];        // bf16 W3 tile, [o][c] (K=c contig)
  alignas(16) unsigned short m_bf[OT * VV * 32];  // m[o][u][v], v padded->32 (K=v contig)
  alignas(16) unsigned short xh_t[400 * CC];      // bf16 x tile [tv][c] (K=c contig)
  alignas(16) unsigned short x3l[OT * 16 * 32];   // x3[o][t16][v], v padded (K=v contig)
};

__global__ __launch_bounds__(256) void ctrgcn_main(
    const unsigned short* __restrict__ xh, const float* __restrict__ xm,
    const float* __restrict__ PA, const float* __restrict__ alpha,
    const float* __restrict__ w1, const float* __restrict__ b1,
    const float* __restrict__ w2, const float* __restrict__ b2,
    const float* __restrict__ w3, const float* __restrict__ b3,
    const float* __restrict__ w4, const float* __restrict__ b4,
    float* __restrict__ y) {
  __shared__ MainLds L;
  int tid = threadIdx.x;
  int lane = tid & 31, wave = tid >> 5;
  int obase = blockIdx.x * OT;
  int s = blockIdx.y;
  int n = blockIdx.z;
  float alpha_s = alpha[s];

  // ---- stage 0: cooperative loads / zero padding ----
  for (int i = tid; i < CC * VV; i += 256) L.xm[i] = xm[(size_t)n * CC * VV + i];
  for (int i = tid; i < VV * VV; i += 256) {
    int u = i / VV, v = i % VV;
    L.pasym[i] = PA[s * VV * VV + u * VV + v] + PA[s * VV * VV + v * VV + u];
  }
  for (int i = tid; i < OT * RR; i += 256)
    L.w4l[i] = w4[(s * OO + obase + i / RR) * RR + (i % RR)];
  if (tid < OT) {
    L.b4l[tid] = b4[s * OO + obase + tid];
    L.b3l[tid] = b3[s * OO + obase + tid];
  }
  for (int i = tid; i < OT * CC; i += 256)
    L.w3l[i] = f2bf(w3[(s * OO + obase + i / CC) * CC + (i % CC)]);
  for (int i = tid; i < OT * VV * 32; i += 256) L.m_bf[i] = 0;
  for (int i = tid; i < OT * 16 * 32; i += 256) L.x3l[i] = 0;
  for (int i = tid; i < OT * VV; i += 256) L.msum[i] = 0.f;
  __syncthreads();

  // ---- stage 1: x1 = W1 @ xm + b1, x2 = W2 @ xm + b2 ----
  for (int task = tid; task < 2 * RR * VV; task += 256) {
    int which = task >= RR * VV;
    int t2 = task - which * RR * VV;
    int r = t2 / VV, u = t2 % VV;
    const float* w = (which ? w2 : w1) + (s * RR + r) * CC;
    float acc = (which ? b2 : b1)[s * RR + r];
    for (int c = 0; c < CC; ++c) acc += w[c] * L.xm[c * VV + u];
    (which ? L.x2 : L.x1)[r * VV + u] = acc;
  }
  __syncthreads();

  // ---- stage 2: att[r,u,v] = tanh(x1[r,u] - x2[r,v]) ----
  for (int i = tid; i < RR * VV * VV; i += 256) {
    int r = i / (VV * VV), rem = i % (VV * VV);
    int u = rem / VV, v = rem % VV;
    L.att[i] = tanhf(L.x1[r * VV + u] - L.x2[r * VV + v]);
  }
  __syncthreads();

  // ---- stage 3: m[o,u,v] = softmax_u( (W4@att + b4)*alpha + Asym ) ----
  for (int task = tid; task < OT * VV; task += 256) {
    int o_l = task / VV, v = task % VV;
    float vals[VV];
    float mx = -3.0e38f;
    #pragma unroll
    for (int u = 0; u < VV; ++u) {
      float acc = 0.f;
      #pragma unroll
      for (int r = 0; r < RR; ++r)
        acc += L.w4l[o_l * RR + r] * L.att[(r * VV + u) * VV + v];
      float mm = (acc + L.b4l[o_l]) * alpha_s + L.pasym[u * VV + v];
      vals[u] = mm;
      mx = fmaxf(mx, mm);
    }
    float sum = 0.f;
    #pragma unroll
    for (int u = 0; u < VV; ++u) { vals[u] = __expf(vals[u] - mx); sum += vals[u]; }
    float inv = 1.f / sum;
    #pragma unroll
    for (int u = 0; u < VV; ++u) {
      float p = vals[u] * inv;
      L.m_bf[(o_l * VV + u) * 32 + v] = f2bf(p);
      atomicAdd(&L.msum[o_l * VV + u], p);
    }
  }
  __syncthreads();

  // ---- stage 4: per t-tile, GEMM1 (x3 = W3 @ X) then GEMM2 (y += x3 @ m^T) --
  for (int tt = 0; tt < TT / 16; ++tt) {
    // load bf16 x tile transposed into [tv][c]: coalesced 32-bit global
    // loads (tv fast axis in memory), two b16 LDS scatters each.
    for (int idx = tid; idx < CC * 200; idx += 256) {
      int c = idx / 200, i = idx - c * 200;              // i-th uint => tv = 2i
      const unsigned int* gp = (const unsigned int*)(
          xh + (((size_t)(n * CC + c) * TT + tt * 16) * VV));
      unsigned int d = gp[i];
      int tv = 2 * i;
      L.xh_t[tv * CC + c]       = (unsigned short)(d & 0xFFFFu);
      L.xh_t[(tv + 1) * CC + c] = (unsigned short)(d >> 16);
    }
    __syncthreads();

    // GEMM1: D[o,tv] = sum_c W3[o,c] * X[c,tv]   (M=16 o, K=64 c, N=400 tv)
    for (int nt = wave; nt < 25; nt += 8) {
      v8f acc = {0.f, 0.f, 0.f, 0.f, 0.f, 0.f, 0.f, 0.f};
      int col = lane & 15;
      int tv = nt * 16 + col;
      #pragma unroll
      for (int ks = 0; ks < 2; ++ks) {
        BF16x16 a, b;
        #pragma unroll
        for (int j = 0; j < 8; ++j) {
          int kb = ks * 32 + kpair(j, lane);
          a.w[j] = *(const unsigned int*)&L.w3l[col * CC + kb];   // A row = o
          b.w[j] = *(const unsigned int*)&L.xh_t[tv * CC + kb];   // B col = tv
        }
        acc = __builtin_amdgcn_wmma_f32_16x16x32_bf16(
            false, a.v, false, b.v, (short)0, acc, false, false);
      }
      int t_l = tv / VV, vv = tv % VV;
      int rb = (lane >= 16) ? 8 : 0;
      #pragma unroll
      for (int j = 0; j < 8; ++j) {
        int o_l = j + rb;                               // D row = o
        L.x3l[(o_l * 16 + t_l) * 32 + vv] = f2bf(acc[j]);
      }
    }
    __syncthreads();

    // GEMM2: per o: D[t,u] = sum_v x3[o,t,v] * m[o,u,v]  (K=32 padded v)
    for (int tile = wave; tile < OT * 2; tile += 8) {
      int o_l = tile >> 1, ut = tile & 1;
      int u = ut * 16 + (lane & 15);
      int uc = (u < VV) ? u : 0;
      BF16x16 a, b;
      #pragma unroll
      for (int j = 0; j < 8; ++j) {
        int vb = kpair(j, lane);
        a.w[j] = *(const unsigned int*)&L.x3l[(o_l * 16 + (lane & 15)) * 32 + vb];
        b.w[j] = *(const unsigned int*)&L.m_bf[(o_l * VV + uc) * 32 + vb];
      }
      v8f acc = {0.f, 0.f, 0.f, 0.f, 0.f, 0.f, 0.f, 0.f};
      acc = __builtin_amdgcn_wmma_f32_16x16x32_bf16(
          false, a.v, false, b.v, (short)0, acc, false, false);
      if (u < VV) {
        float bias = L.b3l[o_l] * L.msum[o_l * VV + u];       // b3 * sum_v m
        int rb = (lane >= 16) ? 8 : 0;
        #pragma unroll
        for (int j = 0; j < 8; ++j) {
          int t_row = tt * 16 + j + rb;                       // D row = t
          size_t oidx = (((size_t)n * OO + obase + o_l) * TT + t_row) * VV + u;
          unsafeAtomicAdd(&y[oidx], acc[j] + bias);
        }
      }
    }
    __syncthreads();
  }
}

// ------------------------------------------------------------------ SE ----
__global__ __launch_bounds__(256) void se_kernel(
    const float* __restrict__ y, const float* __restrict__ bn_w,
    const float* __restrict__ bn_b, const float* __restrict__ se_w1,
    const float* __restrict__ se_b1, const float* __restrict__ se_w2,
    const float* __restrict__ se_b2, float* __restrict__ sev) {
  __shared__ float part[256];
  __shared__ float pooled[OO];
  __shared__ float h[HH];
  int tid = threadIdx.x, n = blockIdx.x;
  int o = tid >> 2, p4 = tid & 3;
  const float* row = y + ((size_t)n * OO + o) * (TT * VV) + p4 * 400;
  float sacc = 0.f;
  for (int i = 0; i < 400; ++i) sacc += row[i];
  part[tid] = sacc;
  __syncthreads();
  if (tid < OO) {
    float tot = part[tid * 4] + part[tid * 4 + 1] + part[tid * 4 + 2] + part[tid * 4 + 3];
    float bnscale = bn_w[tid] * rsqrtf(1.f + BN_EPS);
    pooled[tid] = bnscale * (tot * (1.f / (TT * VV))) + bn_b[tid];
  }
  __syncthreads();
  if (tid < HH) {
    float a = se_b1[tid];
    for (int o2 = 0; o2 < OO; ++o2) a += se_w1[tid * OO + o2] * pooled[o2];
    h[tid] = fmaxf(a, 0.f);
  }
  __syncthreads();
  if (tid < OO) {
    float a = se_b2[tid];
    for (int j = 0; j < HH; ++j) a += se_w2[tid * HH + j] * h[j];
    sev[n * OO + tid] = 1.f / (1.f + __expf(-a));
  }
}

// ------------------------------------------------------------ finalize ----
__global__ __launch_bounds__(256) void finalize_kernel(
    float* y, const float* __restrict__ x, const float* __restrict__ bn_w,
    const float* __restrict__ bn_b, const float* __restrict__ sev, int total) {
  int idx = blockIdx.x * 256 + threadIdx.x;
  if (idx >= total) return;
  int n = idx / (OO * TT * VV);
  int o = (idx / (TT * VV)) % OO;
  float bnscale = bn_w[o] * rsqrtf(1.f + BN_EPS);
  float ybn = bnscale * y[idx] + bn_b[o];
  float v = ybn * sev[n * OO + o] + x[idx];
  y[idx] = fmaxf(v, 0.f);
}

// -------------------------------------------------------------- launch ----
extern "C" void kernel_launch(void* const* d_in, const int* in_sizes, int n_in,
                              void* d_out, int out_size, void* d_ws, size_t ws_size,
                              hipStream_t stream) {
  (void)in_sizes; (void)n_in; (void)ws_size;
  const float* x     = (const float*)d_in[0];
  const float* PA    = (const float*)d_in[1];
  const float* alpha = (const float*)d_in[2];
  const float* w1    = (const float*)d_in[3];
  const float* b1    = (const float*)d_in[4];
  const float* w2    = (const float*)d_in[5];
  const float* b2    = (const float*)d_in[6];
  const float* w3    = (const float*)d_in[7];
  const float* b3    = (const float*)d_in[8];
  const float* w4    = (const float*)d_in[9];
  const float* b4    = (const float*)d_in[10];
  const float* bn_w  = (const float*)d_in[11];
  const float* bn_b  = (const float*)d_in[12];
  const float* se_w1 = (const float*)d_in[13];
  const float* se_b1 = (const float*)d_in[14];
  const float* se_w2 = (const float*)d_in[15];
  const float* se_b2 = (const float*)d_in[16];

  // workspace layout: xh (bf16 x) | xm | se
  unsigned short* xh = (unsigned short*)d_ws;
  float* xm  = (float*)((char*)d_ws + (size_t)NB * CC * TT * VV * 2);
  float* sev = (float*)((char*)d_ws + (size_t)NB * CC * TT * VV * 2
                                     + (size_t)NB * CC * VV * 4);
  float* y = (float*)d_out;

  hipMemsetAsync(d_out, 0, (size_t)out_size * sizeof(float), stream);
  prep_kernel<<<NB * CC, 256, 0, stream>>>(x, xh, xm);
  ctrgcn_main<<<dim3(OO / OT, SS, NB), 256, 0, stream>>>(
      xh, xm, PA, alpha, w1, b1, w2, b2, w3, b3, w4, b4, y);
  se_kernel<<<NB, 256, 0, stream>>>(y, bn_w, bn_b, se_w1, se_b1, se_w2, se_b2, sev);
  const int total = NB * OO * TT * VV;
  finalize_kernel<<<(total + 255) / 256, 256, 0, stream>>>(y, x, bn_w, bn_b, sev, total);
}
